// Attention_7653631722097
// MI455X (gfx1250) — matrix operand order repaired and checked by reading, not compiled, not run on hardware
//
#include <hip/hip_runtime.h>
#include <math.h>

// ---------------------------------------------------------------------------
// Attention block for MI455X (gfx1250, wave32, WMMA).
//
// Shapes: B=8, C=256, H=W=32 (S=1024), HEADS=8, CH=32, HID=256.
// All matmuls run on v_wmma_f32_16x16x32_bf16 (bf16 in, f32 accumulate).
// GEMM kernels are N-blocked 4x: one wave owns a 16x64 output strip so the
// fp32->bf16 A (weight) fragment is built once and reused by 4 WMMAs.
//
// Workspace layout (bytes):
//   [0,            4 MiB)  Xt   : (b, s, c)   bf16  -- x transposed+converted
//   [4 MiB,       16 MiB)  QKV  : (b,h,sel)   bf16  -- sel 0:Q(s,c) 1:K(t,c) 2:V(c,t)
//   [16 MiB,      20 MiB)  O    : (b, s, c')  bf16  -- attention output
// ---------------------------------------------------------------------------

typedef __bf16 bf16_t;
typedef bf16_t v16bf __attribute__((ext_vector_type(16)));
typedef bf16_t v8bf  __attribute__((ext_vector_type(8)));
typedef float  v8f   __attribute__((ext_vector_type(8)));

#define S_LEN   1024
#define C_DIM   256
#define CH_DIM  32
#define HEADS   8
#define BATCH   8

// --- WMMA wrapper -----------------------------------------------------------
__device__ __forceinline__ v8f wmma_bf16(v16bf a, v16bf b, v8f c) {
  // (neg_a, A, neg_b, B, c_mod, C, reuse_a, reuse_b)
  return __builtin_amdgcn_wmma_f32_16x16x32_bf16(false, a, false, b,
                                                 (short)0, c, false, false);
}

// --- Fragment loaders matching CDNA5 ISA VGPR layouts ------------
// A (16x32, 16-bit): lane l: row m=l%16; VGPR0-3 hold K=(l/16)*8+0..7,
// VGPR4-7 hold K=16+(l/16)*8+0..7.  src is row-major (m, k), stride in elems.
__device__ __forceinline__ v16bf load_a_bf16(const bf16_t* src, int stride, int lane) {
  const bf16_t* p = src + (lane & 15) * stride + (lane >> 4) * 8;
  v8bf lo = *(const v8bf*)p;
  v8bf hi = *(const v8bf*)(p + 16);
  v16bf a;
#pragma unroll
  for (int i = 0; i < 8; ++i) { a[i] = lo[i]; a[i + 8] = hi[i]; }
  return a;
}

// Same A layout but fp32 source, converted to bf16 on the fly (for weights).
__device__ __forceinline__ v16bf load_a_f32(const float* src, int stride, int lane) {
  const float* p = src + (lane & 15) * stride + (lane >> 4) * 8;
  v16bf a;
#pragma unroll
  for (int i = 0; i < 8; ++i) {
    a[i]     = (bf16_t)p[i];
    a[i + 8] = (bf16_t)p[16 + i];
  }
  return a;
}

// B (32x16, 16-bit): lane l: col n=l%16; VGPR0-3 hold K=(l/16)*8+0..7,
// VGPR4-7 hold K=16+(l/16)*8+0..7.  src is stored "N-major": element (k,n)
// at src[n*stride + k], so each lane reads two 16-byte chunks.
__device__ __forceinline__ v16bf load_b_bf16(const bf16_t* src, int stride, int lane) {
  const bf16_t* p = src + (lane & 15) * stride + (lane >> 4) * 8;
  v8bf lo = *(const v8bf*)p;
  v8bf hi = *(const v8bf*)(p + 16);
  v16bf b;
#pragma unroll
  for (int i = 0; i < 8; ++i) { b[i] = lo[i]; b[i + 8] = hi[i]; }
  return b;
}

// ---------------------------------------------------------------------------
// Kernel T: x (b,c,s) f32  ->  Xt (b,s,c) bf16   (coalesced reads)
// ---------------------------------------------------------------------------
__global__ __launch_bounds__(256) void k_transpose(const float* __restrict__ x,
                                                   bf16_t* __restrict__ xt) {
  int idx = blockIdx.x * 256 + threadIdx.x;      // < B*C*S = 2^21
  int s = idx & (S_LEN - 1);
  int c = (idx >> 10) & (C_DIM - 1);
  int b = idx >> 18;
  xt[(b << 18) + (s << 8) + c] = (bf16_t)x[idx];
}

// ---------------------------------------------------------------------------
// Kernel A: QKV projection.  Y(768x1024) = Wqkv(768x256) * X(256x1024),
// + bias, SiLU, scatter to Q(s,c) / K(t,c) / V(c,t) bf16 per (b, head).
// One wave per 16x64 strip: A fragment reused by 4 WMMAs per K-step.
// ---------------------------------------------------------------------------
__global__ __launch_bounds__(128) void k_qkv(const float* __restrict__ wq,
                                             const float* __restrict__ bq,
                                             const bf16_t* __restrict__ xt,
                                             bf16_t* __restrict__ qkv) {
  int lane = threadIdx.x & 31;
  int gw   = blockIdx.x * 4 + (threadIdx.x >> 5);   // global wave id
  int b    = gw / 768;                               // 48 o-tiles * 16 s-groups
  int rem  = gw - b * 768;
  int o0   = (rem >> 4) << 4;                        // 0..752
  int s0   = (rem & 15) << 6;                        // 0..960 step 64

  const bf16_t* xb = xt + (b << 18) + (s0 << 8);
  v8f acc[4] = {};
#pragma unroll
  for (int k0 = 0; k0 < 256; k0 += 32) {
    v16bf a = load_a_f32(wq + o0 * 256 + k0, 256, lane);   // built once
#pragma unroll
    for (int j = 0; j < 4; ++j) {
      v16bf bx = load_b_bf16(xb + (j << 12) + k0, 256, lane);  // j*16*256
      acc[j] = wmma_bf16(a, bx, acc[j]);
    }
  }

  // C layout: lane l -> col s; rows o = o0 + (l/16)*8 + r
  int h    = lane >> 4;
  int ob   = o0 + h * 8;
  int head = ob / 96;
  int rr   = ob - head * 96;
  int sel  = rr >> 5;            // 0:q 1:k 2:v
  int cc   = rr & 31;
  bf16_t* base = qkv + ((((b * HEADS + head) * 3 + sel)) << 15);  // *32768

  float bias[8];
#pragma unroll
  for (int r = 0; r < 8; ++r) bias[r] = bq[ob + r];

#pragma unroll
  for (int j = 0; j < 4; ++j) {
    int s = s0 + j * 16 + (lane & 15);
    if (sel < 2) {               // Q,K stored (s, c): 8 contiguous bf16
      v8bf pk;
#pragma unroll
      for (int r = 0; r < 8; ++r) {
        float v = acc[j][r] + bias[r];
        v = v / (1.0f + __expf(-v));         // SiLU
        pk[r] = (bf16_t)v;
      }
      *(v8bf*)(base + s * CH_DIM + cc) = pk;
    } else {                     // V stored (c, t): strided scalar stores
#pragma unroll
      for (int r = 0; r < 8; ++r) {
        float v = acc[j][r] + bias[r];
        v = v / (1.0f + __expf(-v));
        base[((cc + r) << 10) + s] = (bf16_t)v;
      }
    }
  }
}

// ---------------------------------------------------------------------------
// Kernel B: attention.  One workgroup (4 waves) per (b, head, 16-row s-block).
// scores(16x1024) in LDS f32, two-pass softmax, P bf16 in LDS, PV via WMMA,
// cross-wave K-reduction with ds_add_f32 atomics, final scale by 1/rowsum.
// Dynamic LDS = 100992 bytes.
// ---------------------------------------------------------------------------
__global__ __launch_bounds__(128) void k_attn(const bf16_t* __restrict__ qkv,
                                              bf16_t* __restrict__ o) {
  extern __shared__ char smem[];
  float*  sc     = (float*)smem;                    // 16*1024 f32 = 65536 B
  bf16_t* P      = (bf16_t*)(smem + 65536);         // 16*1024 bf16 = 32768 B
  float*  osum   = (float*)(smem + 98304);          // 16*32 f32 = 2048 B
  float*  red    = (float*)(smem + 100352);         // 128 f32
  float*  rowmax = (float*)(smem + 100864);         // 16 f32
  float*  rinv   = (float*)(smem + 100928);         // 16 f32

  int tid  = threadIdx.x;
  int lane = tid & 31;
  int wave = tid >> 5;
  int s0   = (blockIdx.x & 63) << 4;
  int bh   = blockIdx.x >> 6;                       // b*HEADS + head

  const bf16_t* Q = qkv + ((bh * 3 + 0) << 15);
  const bf16_t* K = qkv + ((bh * 3 + 1) << 15);
  const bf16_t* V = qkv + ((bh * 3 + 2) << 15);

  // ---- scores = (Q^T K) / sqrt(CH): one WMMA per 16x16 tile (K=32 full) ----
  const float scale = 0.17677669529663687f;         // 1/sqrt(32)
  v16bf aq = load_a_bf16(Q + s0 * CH_DIM, CH_DIM, lane);   // shared A
  for (int t0 = wave * 256; t0 < wave * 256 + 256; t0 += 16) {
    v16bf bk = load_b_bf16(K + t0 * CH_DIM, CH_DIM, lane);
    v8f c = {};
    c = wmma_bf16(aq, bk, c);
    int t  = t0 + (lane & 15);
    int sr = (lane >> 4) * 8;
#pragma unroll
    for (int r = 0; r < 8; ++r) sc[((sr + r) << 10) + t] = c[r] * scale;
  }
  __syncthreads();

  for (int i = tid; i < 512; i += 128) osum[i] = 0.0f;

  // ---- softmax over t (8 threads per row) ----
  int row = tid >> 3, j = tid & 7;
  const float* srow = sc + (row << 10);
  float pmax = -3.4e38f;
  for (int i = j * 128; i < j * 128 + 128; ++i) pmax = fmaxf(pmax, srow[i]);
  red[tid] = pmax;
  __syncthreads();
  if (j == 0) {
    float m = red[tid];
#pragma unroll
    for (int q = 1; q < 8; ++q) m = fmaxf(m, red[tid + q]);
    rowmax[row] = m;
  }
  __syncthreads();
  float rm = rowmax[row];
  bf16_t* prow = P + (row << 10);
  float psum = 0.0f;
  for (int i = j * 128; i < j * 128 + 128; ++i) {
    float e = __expf(srow[i] - rm);
    psum += e;
    prow[i] = (bf16_t)e;                            // unnormalized probs
  }
  red[tid] = psum;
  __syncthreads();
  if (j == 0) {
    float ssum = 0.0f;
#pragma unroll
    for (int q = 0; q < 8; ++q) ssum += red[tid + q];
    rinv[row] = 1.0f / ssum;
  }
  __syncthreads();

  // ---- O' = P(16x1024) x V^T(1024x32): each wave owns a 256-wide K slice ---
  v8f acc0 = {}, acc1 = {};
  for (int t0 = wave * 256; t0 < wave * 256 + 256; t0 += 32) {
    v16bf ap  = load_a_bf16(P + t0, S_LEN, lane);               // ds_load
    v16bf bv0 = load_b_bf16(V + t0, S_LEN, lane);               // c = 0..15
    v16bf bv1 = load_b_bf16(V + (16 << 10) + t0, S_LEN, lane);  // c = 16..31
    acc0 = wmma_bf16(ap, bv0, acc0);
    acc1 = wmma_bf16(ap, bv1, acc1);
  }
  {
    int n  = lane & 15;
    int sr = (lane >> 4) * 8;
#pragma unroll
    for (int r = 0; r < 8; ++r) {
      atomicAdd(&osum[(sr + r) * CH_DIM + n],      acc0[r]);   // ds_add_f32
      atomicAdd(&osum[(sr + r) * CH_DIM + 16 + n], acc1[r]);
    }
  }
  __syncthreads();

  // ---- normalize + store O[b][s][head*32+c] (bf16) ----
  int head = bh & (HEADS - 1);
  int b    = bh >> 3;
  bf16_t* ob = o + (b << 18) + (head << 5);
  for (int i = tid; i < 512; i += 128) {
    int s = i >> 5, c = i & 31;
    float v = osum[i] * rinv[s];
    ob[((s0 + s) << 8) + c] = (bf16_t)v;
  }
}

// ---------------------------------------------------------------------------
// Kernel C: out = W_out(256x256) * O(256x1024) + b_out + x  (fp32 residual)
// One wave per 16x64 strip: A fragment reused by 4 WMMAs per K-step.
// ---------------------------------------------------------------------------
__global__ __launch_bounds__(128) void k_out(const float* __restrict__ wo,
                                             const float* __restrict__ bo,
                                             const bf16_t* __restrict__ O,
                                             const float* __restrict__ x,
                                             float* __restrict__ out) {
  int lane = threadIdx.x & 31;
  int gw   = blockIdx.x * 4 + (threadIdx.x >> 5);
  int b    = gw >> 8;                   // 16 o-tiles * 16 s-groups per batch
  int rem  = gw & 255;
  int o0   = (rem >> 4) << 4;
  int s0   = (rem & 15) << 6;

  const bf16_t* Ob = O + (b << 18) + (s0 << 8);
  v8f acc[4] = {};
#pragma unroll
  for (int k0 = 0; k0 < 256; k0 += 32) {
    v16bf a = load_a_f32(wo + o0 * 256 + k0, 256, lane);   // built once
#pragma unroll
    for (int j = 0; j < 4; ++j) {
      v16bf bb = load_b_bf16(Ob + (j << 12) + k0, 256, lane);
      acc[j] = wmma_bf16(a, bb, acc[j]);
    }
  }

  int obi = o0 + (lane >> 4) * 8;
  const float* xb = x + (b << 18);
  float* yb = out + (b << 18);
#pragma unroll
  for (int j = 0; j < 4; ++j) {
    int s = s0 + j * 16 + (lane & 15);
#pragma unroll
    for (int r = 0; r < 8; ++r) {
      int oo = obi + r;
      yb[(oo << 10) + s] = acc[j][r] + bo[oo] + xb[(oo << 10) + s];
    }
  }
}

// ---------------------------------------------------------------------------
extern "C" void kernel_launch(void* const* d_in, const int* in_sizes, int n_in,
                              void* d_out, int out_size, void* d_ws, size_t ws_size,
                              hipStream_t stream) {
  const float* x     = (const float*)d_in[0];
  const float* w_qkv = (const float*)d_in[1];
  const float* b_qkv = (const float*)d_in[2];
  const float* w_out = (const float*)d_in[3];
  const float* b_out = (const float*)d_in[4];

  bf16_t* xt   = (bf16_t*)d_ws;
  bf16_t* qkv  = (bf16_t*)((char*)d_ws + (4u  << 20));   //  4 MiB
  bf16_t* Obuf = (bf16_t*)((char*)d_ws + (16u << 20));   // 16 MiB
  float*  out  = (float*)d_out;

  k_transpose<<<8192, 256, 0, stream>>>(x, xt);
  k_qkv<<<1536, 128, 0, stream>>>(w_qkv, b_qkv, xt, qkv);   // 6144 waves
  k_attn<<<4096, 128, 100992, stream>>>(qkv, Obuf);         // 64 bh * 64 s-blk
  k_out<<<512, 128, 0, stream>>>(w_out, b_out, Obuf, x, out); // 2048 waves
}
